// STDIM_9320079032778
// MI455X (gfx1250) — compile-verified, matrix operand order checked by reading, not run on hardware
//
#include <hip/hip_runtime.h>
#include <hip/hip_bf16.h>

typedef __attribute__((ext_vector_type(16))) __bf16 v16bf;
typedef __attribute__((ext_vector_type(8)))  float  v8f;
typedef __attribute__((ext_vector_type(4)))  unsigned int v4u;
typedef __attribute__((ext_vector_type(8)))  int v8i;
typedef __attribute__((ext_vector_type(4)))  int v4i;

#define N_   1024
#define C_   384
#define S_   49
#define D_   512
#define DM_  64
#define PITCH 520          // bf16 elems per LDS row: 512 + 8 pad (1040 B -> conflict-free frag reads)
#define NCH  (N_ / 16)     // 64 column chunks
#define BUFSH (16 * PITCH) // bf16 elems per LDS buffer (16.25 KB)

#if defined(__has_builtin)
#if __has_builtin(__builtin_amdgcn_tensor_load_to_lds)
#define USE_TDM 1
#endif
#if __has_builtin(__builtin_amdgcn_sched_group_barrier)
#define HAVE_SGB 1
#endif
#endif
#ifndef USE_TDM
#define USE_TDM 0
#endif
#ifndef HAVE_SGB
#define HAVE_SGB 0
#endif

__device__ __forceinline__ unsigned short f2bf(float f) {
    unsigned int u = __float_as_uint(f);
    u += 0x7FFFu + ((u >> 16) & 1u);       // round-to-nearest-even
    return (unsigned short)(u >> 16);
}

// ---------------- prep kernels ----------------

__global__ void pack_pos(const float* __restrict__ x, const float* __restrict__ m,
                         const float* __restrict__ c, unsigned short* __restrict__ dst) {
    long long i = (long long)blockIdx.x * blockDim.x + threadIdx.x;
    const long long total = (long long)S_ * N_ * D_;
    if (i >= total) return;
    int d = (int)(i % D_);
    long long t = i / D_;
    int j = (int)(t % N_);
    int s = (int)(t / N_);
    float v;
    if (d < C_)            v = x[(long long)j * (C_ * S_) + (long long)d * S_ + s];
    else if (d < C_ + DM_) v = m[j * DM_ + (d - C_)];
    else                   v = c[j * DM_ + (d - C_ - DM_)];
    dst[i] = f2bf(v);
}

__global__ void conv_f(const float* __restrict__ f, unsigned short* __restrict__ dst) {
    int i = blockIdx.x * blockDim.x + threadIdx.x;
    if (i < N_ * D_) dst[i] = f2bf(f[i]);
}

__global__ void copy_out(const float* __restrict__ a, const float* __restrict__ b,
                         float* __restrict__ out) {
    const long long per = (long long)N_ * C_ * S_;
    long long stride = (long long)gridDim.x * blockDim.x;
    for (long long i = (long long)blockIdx.x * blockDim.x + threadIdx.x; i < per; i += stride) {
        out[1 + i] = a[i];
        out[1 + per + i] = b[i];
    }
}

// ---------------- fragment helpers ----------------

union FragU { uint4 q[2]; v16bf v; };

// ISA 16-bit operand layout: lanes 0-15 -> K {kb+0..7, kb+16..23}, lanes 16-31 -> K {kb+8..15, kb+24..31}
__device__ __forceinline__ v16bf load_frag(const unsigned short* __restrict__ base,
                                           int row0, int kbase, int lane) {
    FragU f;
    int row  = row0 + (lane & 15);
    int coff = (lane & 16) ? 8 : 0;
    const unsigned short* p = base + (long long)row * D_ + kbase + coff;
    f.q[0] = *(const uint4*)(p);
    f.q[1] = *(const uint4*)(p + 16);
    return f.v;
}

__device__ __forceinline__ v16bf lds_frag(const unsigned short* buf, int kbase, int lane) {
    FragU f;
    const unsigned short* p = buf + (lane & 15) * PITCH + kbase + ((lane & 16) ? 8 : 0);
    f.q[0] = *(const uint4*)(p);       // ds_load_b128
    f.q[1] = *(const uint4*)(p + 16);
    return f.v;
}

#if USE_TDM
// Issue one TDM DMA: 16 rows x 512 bf16 tile of pos[s] -> LDS buffer with 16B row pad.
__device__ __forceinline__ void tdm_load_chunk(const unsigned short* gsrc, unsigned lds_off) {
    unsigned long long ga = (unsigned long long)(size_t)gsrc;
    v4u g0;
    g0.x = 1u;                                               // count = 1 (valid descriptor)
    g0.y = lds_off;                                          // lds_addr
    g0.z = (unsigned)ga;                                     // global_addr[31:0]
    g0.w = (unsigned)((ga >> 32) & 0x01FFFFFFull) | (2u << 30); // global_addr[56:32] | type=2
    v8i g1;
    g1[0] = (int)((1u << 16) | (1u << 20) | (7u << 22) | (3u << 25)); // 2B elems, pad 16B per 1024B
    g1[1] = (int)(512u << 16);                               // tensor_dim0 = 512
    g1[2] = (int)(((unsigned)N_ & 0xFFFFu) << 16);           // tensor_dim1 = N
    g1[3] = (int)(512u << 16);                               // tile_dim0 = 512
    g1[4] = 16;                                              // tile_dim1 = 16
    g1[5] = 512;                                             // tensor_dim0_stride lo
    g1[6] = (int)(512u << 16);                               // dim0_stride hi=0 | dim1_stride lo
    g1[7] = 0;
    v4i gz = {0, 0, 0, 0};
#if __clang_major__ >= 23
    v8i gz8 = {0, 0, 0, 0, 0, 0, 0, 0};
    __builtin_amdgcn_tensor_load_to_lds(g0, g1, gz, gz, gz8, 0);
#else
    __builtin_amdgcn_tensor_load_to_lds(g0, g1, gz, gz, 0);
#endif
}
#else
__device__ __forceinline__ uint4 gld(const unsigned short* __restrict__ g, int q) {
    int j = q >> 6, u = q & 63;
    return *(const uint4*)(g + (long long)j * D_ + u * 8);
}
__device__ __forceinline__ void sst(unsigned short* l, int q, uint4 v) {
    int j = q >> 6, u = q & 63;
    *(uint4*)(l + j * PITCH + u * 8) = v;
}
#endif

// ---------------- fused GEMM + CE kernel ----------------
// grid = 2 losses * S * (N/128); block = 256 (8 waves). Wave owns a 16-row A strip
// (resident in 128 VGPRs); B chunks are TDM-DMA'd into double-buffered LDS shared by
// all 8 waves. Inner loop: two independent accumulation chains with a rotating
// two-fragment-ahead LDS prefetch, pinned by sched_group_barrier so ds_loads run
// ahead of the WMMA chain instead of wait-0 serialization.
__global__ void __launch_bounds__(256)
ce_gemm(const unsigned short* __restrict__ P, const unsigned short* __restrict__ Q,
        const unsigned short* __restrict__ F, float* __restrict__ partials) {
    __shared__ unsigned short smem[2][BUFSH];

    int bid = blockIdx.x;
    int lossId = bid & 1; bid >>= 1;
    int s  = bid % S_;
    int rb = bid / S_;
    int tid  = threadIdx.x;
    int wave = tid >> 5;
    int lane = tid & 31;
    int half = (lane >> 4) & 1;
    int m0 = rb * 128 + wave * 16;

    const unsigned short* Bbase = P + (size_t)s * (N_ * D_);
    const unsigned short* Abase = lossId ? (Q + (size_t)s * (N_ * D_)) : F;

    // Stage chunk 0
#if USE_TDM
    if (wave == 0) tdm_load_chunk(Bbase, (unsigned)(size_t)&smem[0][0]);
#else
    {
        #pragma unroll
        for (int k = 0; k < 4; ++k) sst(&smem[0][0], tid + k * 256, gld(Bbase, tid + k * 256));
    }
#endif

    // Resident A strip: 16 fragments x 8 VGPRs
    v16bf a[16];
    #pragma unroll
    for (int kc = 0; kc < 16; ++kc) a[kc] = load_frag(Abase, m0, kc * 32, lane);

#if USE_TDM
    if (wave == 0) __builtin_amdgcn_s_wait_tensorcnt(0);
#endif
    __syncthreads();

    float mrun[8], lrun[8], diag[8];
    #pragma unroll
    for (int r = 0; r < 8; ++r) { mrun[r] = -3.0e38f; lrun[r] = 0.0f; diag[r] = 0.0f; }

    for (int jc = 0; jc < NCH; ++jc) {
        const unsigned short* cur = &smem[jc & 1][0];
        unsigned short* nxt = &smem[(jc + 1) & 1][0];
        bool have = (jc + 1) < NCH;
        const unsigned short* gnext = Bbase + (size_t)(jc + 1) * 16 * D_;

        // Kick off DMA for the next chunk before computing on this one
#if USE_TDM
        if (have && wave == 0) tdm_load_chunk(gnext, (unsigned)(size_t)nxt);
#else
        uint4 t0 = {0,0,0,0}, t1 = {0,0,0,0}, t2 = {0,0,0,0}, t3 = {0,0,0,0};
        if (have) {
            t0 = gld(gnext, tid);
            t1 = gld(gnext, tid + 256);
            t2 = gld(gnext, tid + 512);
            t3 = gld(gnext, tid + 768);
        }
#endif

        // Dual accumulation chains, rotating two-fragment lookahead from LDS.
        v16bf b0 = lds_frag(cur, 0 * 32, lane);
        v16bf b1 = lds_frag(cur, 1 * 32, lane);
        v8f acc0 = {0.f, 0.f, 0.f, 0.f, 0.f, 0.f, 0.f, 0.f};
        v8f acc1 = {0.f, 0.f, 0.f, 0.f, 0.f, 0.f, 0.f, 0.f};
        #pragma unroll
        for (int kc = 0; kc < 16; kc += 2) {
            v16bf p0 = b0, p1 = b1;
            if (kc + 2 < 16) b0 = lds_frag(cur, (kc + 2) * 32, lane);
            if (kc + 3 < 16) b1 = lds_frag(cur, (kc + 3) * 32, lane);
            acc0 = __builtin_amdgcn_wmma_f32_16x16x32_bf16(
                       false, a[kc],     false, p0, (short)0, acc0, false, false);
            acc1 = __builtin_amdgcn_wmma_f32_16x16x32_bf16(
                       false, a[kc + 1], false, p1, (short)0, acc1, false, false);
        }

#if HAVE_SGB
        // Pin the schedule: 4 DS-reads ahead, then {1 WMMA, 2 DS-reads} x14, then 2 WMMAs.
        __builtin_amdgcn_sched_group_barrier(0x100, 4, 0);   // DS read
        #pragma unroll
        for (int i = 0; i < 14; ++i) {
            __builtin_amdgcn_sched_group_barrier(0x008, 1, 0); // MFMA/WMMA
            __builtin_amdgcn_sched_group_barrier(0x100, 2, 0); // DS read
        }
        __builtin_amdgcn_sched_group_barrier(0x008, 1, 0);
        __builtin_amdgcn_sched_group_barrier(0x008, 1, 0);
#endif

        v8f acc;
        #pragma unroll
        for (int r = 0; r < 8; ++r) acc[r] = acc0[r] + acc1[r];

        // Diagonal tile: logits[i][i] for this strip's rows
        if (jc * 16 == m0) {
            #pragma unroll
            for (int r = 0; r < 8; ++r) {
                int src = half ? (24 + r) : r;   // C layout: VGPR r = row r(+8), lane%16 = col
                diag[r] = __shfl(acc[r], src, 32);
            }
        }

        // Per-lane online logsumexp (no cross-lane traffic in the loop)
        #pragma unroll
        for (int r = 0; r < 8; ++r) {
            float v  = acc[r];
            float nm = fmaxf(mrun[r], v);
            lrun[r] = lrun[r] * __expf(mrun[r] - nm) + __expf(v - nm);
            mrun[r] = nm;
        }

#if USE_TDM
        if (wave == 0) __builtin_amdgcn_s_wait_tensorcnt(0);
#else
        if (have) {
            sst(nxt, tid,       t0);
            sst(nxt, tid + 256, t1);
            sst(nxt, tid + 512, t2);
            sst(nxt, tid + 768, t3);
        }
#endif
        __syncthreads();
    }

    // Merge (m,l) across the 16-lane column group, then finalize rows
    float rsum = 0.0f;
    #pragma unroll
    for (int r = 0; r < 8; ++r) {
        float m = mrun[r], l = lrun[r];
        #pragma unroll
        for (int msk = 1; msk < 16; msk <<= 1) {
            float mo = __shfl_xor(m, msk, 32);
            float lo = __shfl_xor(l, msk, 32);
            float nm = fmaxf(m, mo);
            l = l * __expf(m - nm) + lo * __expf(mo - nm);
            m = nm;
        }
        rsum += m + __logf(l) - diag[r];
    }
    float o0  = __shfl(rsum, 0, 32);    // rows 0..7 of strip
    float o16 = __shfl(rsum, 16, 32);   // rows 8..15 of strip
    if (lane == 0) partials[blockIdx.x * 8 + wave] = o0 + o16;
}

// Deterministic fixed-order reduction of per-wave partials -> out[0]
__global__ void reduce_loss(const float* __restrict__ part, float* __restrict__ out, int n) {
    __shared__ float sm[256];
    float t = 0.0f;
    for (int i = threadIdx.x; i < n; i += 256) t += part[i];
    sm[threadIdx.x] = t;
    __syncthreads();
    for (int w = 128; w > 0; w >>= 1) {
        if (threadIdx.x < w) sm[threadIdx.x] += sm[threadIdx.x + w];
        __syncthreads();
    }
    if (threadIdx.x == 0) out[0] = sm[0] * (1.0f / (float)(S_ * N_));
}

extern "C" void kernel_launch(void* const* d_in, const int* in_sizes, int n_in,
                              void* d_out, int out_size, void* d_ws, size_t ws_size,
                              hipStream_t stream) {
    const float* f  = (const float*)d_in[0];
    const float* xl = (const float*)d_in[1];
    const float* xp = (const float*)d_in[2];
    const float* mt = (const float*)d_in[3];
    const float* mp = (const float*)d_in[4];
    const float* ct = (const float*)d_in[5];
    const float* cp = (const float*)d_in[6];
    float* out = (float*)d_out;

    const long long pq = (long long)S_ * N_ * D_;
    unsigned short* P  = (unsigned short*)d_ws;
    unsigned short* Qm = P + pq;
    unsigned short* Fb = Qm + pq;
    float* parts = (float*)(Fb + (long long)N_ * D_);

    const int thr = 256;
    int blocksPQ = (int)((pq + thr - 1) / thr);
    pack_pos<<<blocksPQ, thr, 0, stream>>>(xp, mp, cp, P);
    pack_pos<<<blocksPQ, thr, 0, stream>>>(xl, mt, ct, Qm);
    conv_f<<<(N_ * D_ + thr - 1) / thr, thr, 0, stream>>>(f, Fb);
    copy_out<<<2048, thr, 0, stream>>>(xp, xl, out);

    int gblocks = 2 * S_ * (N_ / 128);   // 784 workgroups
    ce_gemm<<<gblocks, 256, 0, stream>>>(P, Qm, Fb, parts);
    reduce_loss<<<1, 256, 0, stream>>>(parts, out, gblocks * 8);
}
